// TransformerEncoderLayer_67276367725150
// MI455X (gfx1250) — matrix operand order repaired and checked by reading, not compiled, not run on hardware
//
#include <hip/hip_runtime.h>
#include <hip/hip_bf16.h>

// ---------------------------------------------------------------------------
// Types for CDNA5 WMMA (wave32): v_wmma_f32_16x16x32_bf16
// ---------------------------------------------------------------------------
typedef __bf16 bf16_t;
typedef __attribute__((ext_vector_type(16))) __bf16 v16bf;
typedef __attribute__((ext_vector_type(8)))  __bf16 v8bf;
typedef __attribute__((ext_vector_type(8)))  float  v8f;

// Problem dimensions (fixed by the reference)
constexpr int S_  = 2048;
constexpr int B_  = 2;
constexpr int D_  = 512;
constexpr int H_  = 8;
constexpr int F_  = 2048;
constexpr int DK_ = 64;

// ---------------------------------------------------------------------------
// Async global -> LDS staging (CDNA5 Tensor/async path, ASYNCcnt-tracked).
// One 16-byte chunk per thread; 256 threads fill a 64x32 bf16 tile (4 KB).
// ---------------------------------------------------------------------------
__device__ __forceinline__ void stage_async_b128(const bf16_t* gsrc, bf16_t* lds_dst) {
  const uint64_t ga = (uint64_t)(uintptr_t)gsrc;
  const uint32_t la = (uint32_t)(uintptr_t)lds_dst;   // low 32 bits = LDS offset
  asm volatile("global_load_async_to_lds_b128 %0, %1, off"
               :: "v"(la), "v"(ga) : "memory");
}

__device__ __forceinline__ void wait_asynccnt0() {
  asm volatile("s_wait_asynccnt 0" ::: "memory");
}

// ---------------------------------------------------------------------------
// Fragment loaders (bf16 A 16x32 / B 32x16 per ISA VGPR layouts)
// ---------------------------------------------------------------------------
__device__ __forceinline__ v16bf load_frag_a_bf16(const bf16_t* __restrict__ A,
                                                  int lda, int m0, int k0, int lane) {
  const int row = m0 + (lane & 15);
  const int kh  = (lane >> 4) * 8;          // 0 or 8
  const bf16_t* p = A + (size_t)row * lda + k0 + kh;
  union { v16bf v; v8bf h[2]; } u;
  u.h[0] = *(const v8bf*)(p);
  u.h[1] = *(const v8bf*)(p + 16);
  return u.v;
}

// Same fragment but source is fp32 (softmax probabilities), converted to bf16.
__device__ __forceinline__ v16bf load_frag_a_f32(const float* __restrict__ A,
                                                 int lda, int m0, int k0, int lane) {
  const int row = m0 + (lane & 15);
  const int kh  = (lane >> 4) * 8;
  const float* p = A + (size_t)row * lda + k0 + kh;
  v8f c0 = *(const v8f*)(p);
  v8f c1 = *(const v8f*)(p + 16);
  v16bf r;
#pragma unroll
  for (int j = 0; j < 8; ++j) r[j]     = (bf16_t)c0[j];
#pragma unroll
  for (int j = 0; j < 8; ++j) r[8 + j] = (bf16_t)c1[j];
  return r;
}

// B tile stored in LDS as [n][k], k-stride 32 bf16: two 16-byte chunks per lane (k = kg..kg+7, 16+kg..16+kg+7).
__device__ __forceinline__ v16bf load_frag_b_lds(const bf16_t* __restrict__ Bs,
                                                 int n_sub, int lane) {
  const int n  = n_sub + (lane & 15);
  const int kg = (lane >> 4) * 8;
  union { v16bf v; v8bf h[2]; } u;
  u.h[0] = *(const v8bf*)(Bs + n * 32 + kg);
  u.h[1] = *(const v8bf*)(Bs + n * 32 + 16 + kg);
  return u.v;
}

__device__ __forceinline__ v8f wmma_bf16(v16bf a, v16bf b, v8f c) {
  return __builtin_amdgcn_wmma_f32_16x16x32_bf16(false, a, false, b, (short)0, c,
                                                 false, false);
}

// ---------------------------------------------------------------------------
// Block reductions (wave32)
// ---------------------------------------------------------------------------
__device__ __forceinline__ float block_reduce_sum(float v, float* sh) {
  const int lane = threadIdx.x & 31, w = threadIdx.x >> 5;
#pragma unroll
  for (int o = 16; o > 0; o >>= 1) v += __shfl_xor(v, o, 32);
  if (lane == 0) sh[w] = v;
  __syncthreads();
  if (w == 0) {
    float r = (lane < 8) ? sh[lane] : 0.0f;
#pragma unroll
    for (int o = 16; o > 0; o >>= 1) r += __shfl_xor(r, o, 32);
    if (lane == 0) sh[0] = r;
  }
  __syncthreads();
  float res = sh[0];
  __syncthreads();
  return res;
}

// ---------------------------------------------------------------------------
// Generic bf16 WMMA GEMM:  C[M,N] = A(M,K) * W(N,K)^T   (i.e. x @ w.T)
// 256 threads = 8 waves; workgroup tile 256(M) x 64(N); K step 32.
// Each wave: 2 A-frags (32 M rows) x 4 B-subtiles = 8 accumulators.
// W tile double-buffered in LDS, filled by async global->LDS copies.
// MODE 0: fp32 row-major out (+optional bias)
// MODE 1: bf16 row-major out, relu(acc + bias)
// MODE 2: bf16 out scattered to (B,H,S,DK)   (Q / K projections)
// MODE 3: bf16 out scattered to (B,H,DK,S)   (V projection, transposed)
// ---------------------------------------------------------------------------
template <int MODE>
__global__ __launch_bounds__(256) void gemm_bf16_kernel(
    const bf16_t* __restrict__ A, const bf16_t* __restrict__ W,
    void* __restrict__ dst, const float* __restrict__ bias,
    int M, int N, int K, int Sdim, int DKdim) {
  __shared__ __align__(32) bf16_t Bs[2][64 * 32];

  const int lane = threadIdx.x & 31;
  const int wave = threadIdx.x >> 5;
  const int n0 = blockIdx.x * 64;
  const int m0 = blockIdx.y * 256 + wave * 32;

  const int t  = threadIdx.x;
  const int nl = t >> 2;            // 0..63
  const int kc = (t & 3) * 8;       // 0,8,16,24
  const int nk = K >> 5;

  v8f acc[8] = {};

  // prologue: stage tile 0, preload A-frags for k0 = 0
  stage_async_b128(&W[(size_t)(n0 + nl) * K + kc], &Bs[0][nl * 32 + kc]);
  v16bf a0 = load_frag_a_bf16(A, K, m0,      0, lane);
  v16bf a1 = load_frag_a_bf16(A, K, m0 + 16, 0, lane);

  for (int ks = 0; ks < nk; ++ks) {
    const int cur = ks & 1;
    wait_asynccnt0();       // this wave's async fill of Bs[cur] is done
    __syncthreads();        // everyone's fill visible; prev compute finished

    if (ks + 1 < nk)
      stage_async_b128(&W[(size_t)(n0 + nl) * K + (ks + 1) * 32 + kc],
                       &Bs[cur ^ 1][nl * 32 + kc]);

    v16bf na0 = a0, na1 = a1;
    if (ks + 1 < nk) {
      na0 = load_frag_a_bf16(A, K, m0,      (ks + 1) * 32, lane);
      na1 = load_frag_a_bf16(A, K, m0 + 16, (ks + 1) * 32, lane);
    }

    const bf16_t* Bsc = &Bs[cur][0];
#pragma unroll
    for (int j = 0; j < 4; ++j) {
      v16bf b = load_frag_b_lds(Bsc, j * 16, lane);
      acc[j]     = wmma_bf16(a0, b, acc[j]);
      acc[j + 4] = wmma_bf16(a1, b, acc[j + 4]);
    }
    a0 = na0; a1 = na1;
  }

  // C/D layout: elem i of v8f -> M = i + (lane>=16 ? 8:0), N = lane&15
  const int lanelo = lane & 15;
  const int mh = (lane >> 4) * 8;
#pragma unroll
  for (int p = 0; p < 2; ++p) {
#pragma unroll
    for (int j = 0; j < 4; ++j) {
      const int n = n0 + j * 16 + lanelo;
#pragma unroll
      for (int i = 0; i < 8; ++i) {
        const int m = m0 + p * 16 + mh + i;
        float c = acc[p * 4 + j][i];
        if (MODE == 0) {
          ((float*)dst)[(size_t)m * N + n] = c + (bias ? bias[n] : 0.0f);
        } else if (MODE == 1) {
          ((bf16_t*)dst)[(size_t)m * N + n] = (bf16_t)fmaxf(c + bias[n], 0.0f);
        } else if (MODE == 2) {
          const int b = m / Sdim, s = m % Sdim;
          const int h = n / DKdim, dk = n % DKdim;
          const int Hn = N / DKdim;
          ((bf16_t*)dst)[(((size_t)(b * Hn + h)) * Sdim + s) * DKdim + dk] = (bf16_t)c;
        } else if (MODE == 3) {
          const int b = m / Sdim, s = m % Sdim;
          const int h = n / DKdim, dk = n % DKdim;
          const int Hn = N / DKdim;
          ((bf16_t*)dst)[(((size_t)(b * Hn + h)) * DKdim + dk) * Sdim + s] = (bf16_t)c;
        }
      }
    }
  }
}

// ---------------------------------------------------------------------------
// Attention scores per (b,h):  scores[q,k] = 0.125 * (q . k) * sph[q,k]
// Written (pre-softmax) straight into the p_attn output region.
// ---------------------------------------------------------------------------
__global__ __launch_bounds__(256) void attn_scores_kernel(
    const bf16_t* __restrict__ q, const bf16_t* __restrict__ kmat,
    const float* __restrict__ sph, float* __restrict__ scores) {
  __shared__ __align__(32) bf16_t Bs[2][64 * 32];

  const int z = blockIdx.z;                          // b*H + h
  const bf16_t* qz = q    + (size_t)z * S_ * DK_;
  const bf16_t* kz = kmat + (size_t)z * S_ * DK_;
  const float* sphz = sph + (size_t)z * S_ * S_;
  float* outz = scores + (size_t)z * S_ * S_;

  const int lane = threadIdx.x & 31;
  const int wave = threadIdx.x >> 5;
  const int n0 = blockIdx.x * 64;
  const int m0 = blockIdx.y * 256 + wave * 32;

  const int t = threadIdx.x, nl = t >> 2, kc = (t & 3) * 8;
  const int nk = DK_ >> 5;   // 2

  v8f acc[8] = {};
  stage_async_b128(&kz[(size_t)(n0 + nl) * DK_ + kc], &Bs[0][nl * 32 + kc]);
  v16bf a0 = load_frag_a_bf16(qz, DK_, m0,      0, lane);
  v16bf a1 = load_frag_a_bf16(qz, DK_, m0 + 16, 0, lane);

  for (int ks = 0; ks < nk; ++ks) {
    const int cur = ks & 1;
    wait_asynccnt0();
    __syncthreads();
    if (ks + 1 < nk)
      stage_async_b128(&kz[(size_t)(n0 + nl) * DK_ + (ks + 1) * 32 + kc],
                       &Bs[cur ^ 1][nl * 32 + kc]);
    v16bf na0 = a0, na1 = a1;
    if (ks + 1 < nk) {
      na0 = load_frag_a_bf16(qz, DK_, m0,      (ks + 1) * 32, lane);
      na1 = load_frag_a_bf16(qz, DK_, m0 + 16, (ks + 1) * 32, lane);
    }
    const bf16_t* Bsc = &Bs[cur][0];
#pragma unroll
    for (int j = 0; j < 4; ++j) {
      v16bf b = load_frag_b_lds(Bsc, j * 16, lane);
      acc[j]     = wmma_bf16(a0, b, acc[j]);
      acc[j + 4] = wmma_bf16(a1, b, acc[j + 4]);
    }
    a0 = na0; a1 = na1;
  }

  const int lanelo = lane & 15, mh = (lane >> 4) * 8;
#pragma unroll
  for (int p = 0; p < 2; ++p) {
#pragma unroll
    for (int j = 0; j < 4; ++j) {
      const int n = n0 + j * 16 + lanelo;
#pragma unroll
      for (int i = 0; i < 8; ++i) {
        const int m = m0 + p * 16 + mh + i;
        const size_t idx = (size_t)m * S_ + n;
        outz[idx] = 0.125f * acc[p * 4 + j][i] * sphz[idx];
      }
    }
  }
}

// ---------------------------------------------------------------------------
// Row softmax, in place (rows of 2048, one 256-thread block per row).
// ---------------------------------------------------------------------------
__global__ __launch_bounds__(256) void softmax_rows_kernel(float* __restrict__ data) {
  __shared__ float sh[8];
  float* row = data + (size_t)blockIdx.x * S_;
  const int t = threadIdx.x, lane = t & 31, w = t >> 5;

  float v[8];
  float mx = -3.402823466e38f;
#pragma unroll
  for (int i = 0; i < 8; ++i) { v[i] = row[t + i * 256]; mx = fmaxf(mx, v[i]); }
#pragma unroll
  for (int o = 16; o > 0; o >>= 1) mx = fmaxf(mx, __shfl_xor(mx, o, 32));
  if (lane == 0) sh[w] = mx;
  __syncthreads();
  if (w == 0) {
    float r = (lane < 8) ? sh[lane] : -3.402823466e38f;
#pragma unroll
    for (int o = 16; o > 0; o >>= 1) r = fmaxf(r, __shfl_xor(r, o, 32));
    if (lane == 0) sh[0] = r;
  }
  __syncthreads();
  mx = sh[0];
  __syncthreads();

  float sum = 0.0f;
#pragma unroll
  for (int i = 0; i < 8; ++i) { v[i] = expf(v[i] - mx); sum += v[i]; }
  sum = block_reduce_sum(sum, sh);
  const float inv = 1.0f / sum;
#pragma unroll
  for (int i = 0; i < 8; ++i) row[t + i * 256] = v[i] * inv;
}

// ---------------------------------------------------------------------------
// o = P @ V per (b,h):  A = p (fp32 -> bf16 on load, 2048x2048),
// W = vT (64 x 2048, K-contiguous).  Output bf16 into (B,S,D).
// ---------------------------------------------------------------------------
__global__ __launch_bounds__(256) void attn_pv_kernel(
    const float* __restrict__ p, const bf16_t* __restrict__ vT,
    bf16_t* __restrict__ o) {
  __shared__ __align__(32) bf16_t Bs[2][64 * 32];

  const int z = blockIdx.z;
  const int b = z / H_, h = z % H_;
  const float* pz = p + (size_t)z * S_ * S_;
  const bf16_t* vz = vT + (size_t)z * DK_ * S_;

  const int lane = threadIdx.x & 31;
  const int wave = threadIdx.x >> 5;
  const int m0 = blockIdx.y * 256 + wave * 32;   // N-tile is the whole DK=64

  const int t = threadIdx.x, nl = t >> 2, kc = (t & 3) * 8;
  const int nk = S_ >> 5;   // 64

  v8f acc[8] = {};
  stage_async_b128(&vz[(size_t)nl * S_ + kc], &Bs[0][nl * 32 + kc]);
  v16bf a0 = load_frag_a_f32(pz, S_, m0,      0, lane);
  v16bf a1 = load_frag_a_f32(pz, S_, m0 + 16, 0, lane);

  for (int ks = 0; ks < nk; ++ks) {
    const int cur = ks & 1;
    wait_asynccnt0();
    __syncthreads();
    if (ks + 1 < nk)
      stage_async_b128(&vz[(size_t)nl * S_ + (ks + 1) * 32 + kc],
                       &Bs[cur ^ 1][nl * 32 + kc]);
    v16bf na0 = a0, na1 = a1;
    if (ks + 1 < nk) {
      na0 = load_frag_a_f32(pz, S_, m0,      (ks + 1) * 32, lane);
      na1 = load_frag_a_f32(pz, S_, m0 + 16, (ks + 1) * 32, lane);
    }
    const bf16_t* Bsc = &Bs[cur][0];
#pragma unroll
    for (int j = 0; j < 4; ++j) {
      v16bf bb = load_frag_b_lds(Bsc, j * 16, lane);
      acc[j]     = wmma_bf16(a0, bb, acc[j]);
      acc[j + 4] = wmma_bf16(a1, bb, acc[j + 4]);
    }
    a0 = na0; a1 = na1;
  }

  const int lanelo = lane & 15, mh = (lane >> 4) * 8;
#pragma unroll
  for (int pp = 0; pp < 2; ++pp) {
#pragma unroll
    for (int j = 0; j < 4; ++j) {
      const int n = j * 16 + lanelo;                 // dk
#pragma unroll
      for (int i = 0; i < 8; ++i) {
        const int m = m0 + pp * 16 + mh + i;         // sequence position
        o[((size_t)(b * S_ + m)) * D_ + h * DK_ + n] = (bf16_t)acc[pp * 4 + j][i];
      }
    }
  }
}

// ---------------------------------------------------------------------------
// Fused: src2 = LN(fc_out + residual, g_attn, 1e-6); h = LN(src + src2, g1, 1e-5)
// One 256-thread block per (s,b) row of D=512; handles (B,S,D)<->(S,B,D).
// ---------------------------------------------------------------------------
__global__ __launch_bounds__(256) void ln_attn_kernel(
    const float* __restrict__ fc, const float* __restrict__ src,
    const float* __restrict__ g_attn, const float* __restrict__ b_attn,
    const float* __restrict__ g1, const float* __restrict__ bn1,
    float* __restrict__ h32, bf16_t* __restrict__ hbf) {
  __shared__ float sh[8];
  const int r = blockIdx.x;               // r = s*B + b
  const int b = r % B_, s = r / B_;
  const size_t rA = (size_t)(b * S_ + s) * D_;
  const size_t rE = (size_t)r * D_;
  const int d0 = threadIdx.x, d1 = threadIdx.x + 256;

  float t0a = fc[rA + d0] + src[rE + d0];
  float t0b = fc[rA + d1] + src[rE + d1];
  float mu = block_reduce_sum(t0a + t0b, sh) * (1.0f / D_);
  float va = block_reduce_sum((t0a - mu) * (t0a - mu) + (t0b - mu) * (t0b - mu), sh)
             * (1.0f / D_);
  float rstd = rsqrtf(va + 1e-6f);
  float s2a = (t0a - mu) * rstd * g_attn[d0] + b_attn[d0];
  float s2b = (t0b - mu) * rstd * g_attn[d1] + b_attn[d1];

  float t1a = src[rE + d0] + s2a;
  float t1b = src[rE + d1] + s2b;
  float mu1 = block_reduce_sum(t1a + t1b, sh) * (1.0f / D_);
  float va1 = block_reduce_sum((t1a - mu1) * (t1a - mu1) + (t1b - mu1) * (t1b - mu1), sh)
              * (1.0f / D_);
  float rstd1 = rsqrtf(va1 + 1e-5f);
  float ha = (t1a - mu1) * rstd1 * g1[d0] + bn1[d0];
  float hb = (t1b - mu1) * rstd1 * g1[d1] + bn1[d1];

  h32[rE + d0] = ha; h32[rE + d1] = hb;
  hbf[rE + d0] = (bf16_t)ha; hbf[rE + d1] = (bf16_t)hb;
}

// out = LN(h + ff, g2, bn2, 1e-5)
__global__ __launch_bounds__(256) void ln_out_kernel(
    const float* __restrict__ h, const float* __restrict__ ff,
    const float* __restrict__ g2, const float* __restrict__ bn2,
    float* __restrict__ out) {
  __shared__ float sh[8];
  const size_t rE = (size_t)blockIdx.x * D_;
  const int d0 = threadIdx.x, d1 = threadIdx.x + 256;
  float ta = h[rE + d0] + ff[rE + d0];
  float tb = h[rE + d1] + ff[rE + d1];
  float mu = block_reduce_sum(ta + tb, sh) * (1.0f / D_);
  float va = block_reduce_sum((ta - mu) * (ta - mu) + (tb - mu) * (tb - mu), sh)
             * (1.0f / D_);
  float rstd = rsqrtf(va + 1e-5f);
  out[rE + d0] = (ta - mu) * rstd * g2[d0] + bn2[d0];
  out[rE + d1] = (tb - mu) * rstd * g2[d1] + bn2[d1];
}

// ---------------------------------------------------------------------------
// Prep kernels
// ---------------------------------------------------------------------------
__global__ void f32_to_bf16_kernel(const float* __restrict__ in,
                                   bf16_t* __restrict__ out, int n) {
  for (int i = blockIdx.x * blockDim.x + threadIdx.x; i < n;
       i += gridDim.x * blockDim.x)
    out[i] = (bf16_t)in[i];
}

// x[(b*S+s)*D + d] = src[(s*B+b)*D + d]
__global__ void transpose_src_kernel(const float* __restrict__ src,
                                     bf16_t* __restrict__ x) {
  int i = blockIdx.x * blockDim.x + threadIdx.x;          // over B*S*D
  int d = i & (D_ - 1);
  int r = i >> 9;                                         // b*S + s
  int b = r >> 11, s = r & (S_ - 1);
  x[i] = (bf16_t)src[((size_t)s * B_ + b) * D_ + d];
}

// ---------------------------------------------------------------------------
// Launch
// ---------------------------------------------------------------------------
extern "C" void kernel_launch(void* const* d_in, const int* in_sizes, int n_in,
                              void* d_out, int out_size, void* d_ws, size_t ws_size,
                              hipStream_t stream) {
  const float* src    = (const float*)d_in[0];
  const float* sph    = (const float*)d_in[1];
  const float* w_qs   = (const float*)d_in[2];
  const float* w_ks   = (const float*)d_in[3];
  const float* w_vs   = (const float*)d_in[4];
  const float* w_fc   = (const float*)d_in[5];
  const float* g_attn = (const float*)d_in[6];
  const float* b_attn = (const float*)d_in[7];
  const float* w1     = (const float*)d_in[8];
  const float* b1     = (const float*)d_in[9];
  const float* w2     = (const float*)d_in[10];
  const float* b2     = (const float*)d_in[11];
  const float* g1     = (const float*)d_in[12];
  const float* bn1    = (const float*)d_in[13];
  const float* g2     = (const float*)d_in[14];
  const float* bn2    = (const float*)d_in[15];

  float* out   = (float*)d_out;
  float* pattn = out + (size_t)S_ * B_ * D_;        // second tuple output

  char* wsp = (char*)d_ws;
  auto carve = [&](size_t elems, size_t esz) -> void* {
    void* p = (void*)wsp;
    wsp += (elems * esz + 255) & ~(size_t)255;
    return p;
  };
  bf16_t* x_bf   = (bf16_t*)carve((size_t)B_ * S_ * D_, 2);
  bf16_t* wq_bf  = (bf16_t*)carve((size_t)D_ * D_, 2);
  bf16_t* wk_bf  = (bf16_t*)carve((size_t)D_ * D_, 2);
  bf16_t* wv_bf  = (bf16_t*)carve((size_t)D_ * D_, 2);
  bf16_t* wfc_bf = (bf16_t*)carve((size_t)D_ * D_, 2);
  bf16_t* w1_bf  = (bf16_t*)carve((size_t)F_ * D_, 2);
  bf16_t* w2_bf  = (bf16_t*)carve((size_t)D_ * F_, 2);
  bf16_t* q_bf   = (bf16_t*)carve((size_t)B_ * S_ * D_, 2);
  bf16_t* k_bf   = (bf16_t*)carve((size_t)B_ * S_ * D_, 2);
  bf16_t* vT_bf  = (bf16_t*)carve((size_t)B_ * S_ * D_, 2);
  bf16_t* o_bf   = (bf16_t*)carve((size_t)B_ * S_ * D_, 2);
  float*  fc_f   = (float*)carve((size_t)B_ * S_ * D_, 4);
  float*  h_f    = (float*)carve((size_t)S_ * B_ * D_, 4);
  bf16_t* h_bf   = (bf16_t*)carve((size_t)S_ * B_ * D_, 2);
  bf16_t* a1_bf  = (bf16_t*)carve((size_t)S_ * B_ * F_, 2);
  float*  ff_f   = (float*)carve((size_t)S_ * B_ * D_, 4);

  const int M  = B_ * S_;   // 4096 rows for projections / FFN

  // 1. prep: transpose + weight conversion
  transpose_src_kernel<<<(B_ * S_ * D_) / 256, 256, 0, stream>>>(src, x_bf);
  f32_to_bf16_kernel<<<512, 256, 0, stream>>>(w_qs, wq_bf, D_ * D_);
  f32_to_bf16_kernel<<<512, 256, 0, stream>>>(w_ks, wk_bf, D_ * D_);
  f32_to_bf16_kernel<<<512, 256, 0, stream>>>(w_vs, wv_bf, D_ * D_);
  f32_to_bf16_kernel<<<512, 256, 0, stream>>>(w_fc, wfc_bf, D_ * D_);
  f32_to_bf16_kernel<<<1024, 256, 0, stream>>>(w1, w1_bf, F_ * D_);
  f32_to_bf16_kernel<<<1024, 256, 0, stream>>>(w2, w2_bf, D_ * F_);

  // 2. Q/K/V projections (WMMA)
  dim3 gQKV(D_ / 64, M / 256, 1);
  gemm_bf16_kernel<2><<<gQKV, 256, 0, stream>>>(x_bf, wq_bf, q_bf, nullptr,
                                                M, D_, D_, S_, DK_);
  gemm_bf16_kernel<2><<<gQKV, 256, 0, stream>>>(x_bf, wk_bf, k_bf, nullptr,
                                                M, D_, D_, S_, DK_);
  gemm_bf16_kernel<3><<<gQKV, 256, 0, stream>>>(x_bf, wv_bf, vT_bf, nullptr,
                                                M, D_, D_, S_, DK_);

  // 3. scores = 0.125 * (q k^T) * sph  -> p_attn region (pre-softmax)
  dim3 gSc(S_ / 64, S_ / 256, B_ * H_);
  attn_scores_kernel<<<gSc, 256, 0, stream>>>(q_bf, k_bf, sph, pattn);

  // 4. softmax in place (this IS the p_attn output)
  softmax_rows_kernel<<<B_ * H_ * S_, 256, 0, stream>>>(pattn);

  // 5. o = P @ V
  dim3 gPV(1, S_ / 256, B_ * H_);
  attn_pv_kernel<<<gPV, 256, 0, stream>>>(pattn, vT_bf, o_bf);

  // 6. fc projection (fp32 out, no bias)
  dim3 gFc(D_ / 64, M / 256, 1);
  gemm_bf16_kernel<0><<<gFc, 256, 0, stream>>>(o_bf, wfc_bf, fc_f, nullptr,
                                               M, D_, D_, S_, DK_);

  // 7. residual + LN(attn) + residual + LN1 -> h (fp32 + bf16)
  ln_attn_kernel<<<S_ * B_, 256, 0, stream>>>(fc_f, src, g_attn, b_attn,
                                              g1, bn1, h_f, h_bf);

  // 8. FFN
  dim3 gF1(F_ / 64, M / 256, 1);
  gemm_bf16_kernel<1><<<gF1, 256, 0, stream>>>(h_bf, w1_bf, a1_bf, b1,
                                               M, F_, D_, S_, DK_);
  dim3 gF2(D_ / 64, M / 256, 1);
  gemm_bf16_kernel<0><<<gF2, 256, 0, stream>>>(a1_bf, w2_bf, ff_f, b2,
                                               M, D_, F_, S_, DK_);

  // 9. final residual + LN2 -> out
  ln_out_kernel<<<S_ * B_, 256, 0, stream>>>(h_f, ff_f, g2, bn2, out);

  (void)in_sizes; (void)n_in; (void)out_size; (void)ws_size;
}
